// TextureWarpingModule_83124797047602
// MI455X (gfx1250) — compile-verified, matrix operand order checked by reading, not run on hardware
//
#include <hip/hip_runtime.h>
#include <hip/hip_bf16.h>

// ---------------------------------------------------------------------------
// CDNA5 (gfx1250): TextureWarpingModule forward.
// All GEMM-shaped convs via v_wmma_f32_16x16x32_bf16 implicit GEMM
// (64x128 block tile, 8 wave32, 4 WMMA/wave/K-step, double-buffered LDS).
// Fixed problem dims baked in: B=4, C=256, H=W=64, HW=4096, N=16384.
// ---------------------------------------------------------------------------

typedef __attribute__((ext_vector_type(16))) __bf16 v16bf;
typedef __attribute__((ext_vector_type(8)))  float  v8f;

__device__ __forceinline__ unsigned short f2bf(float f) {
  unsigned u = __float_as_uint(f);
  u += 0x7FFFu + ((u >> 16) & 1u);      // round-to-nearest-even
  return (unsigned short)(u >> 16);
}
__device__ __forceinline__ unsigned pack2(float a, float b) {
  return (unsigned)f2bf(a) | ((unsigned)f2bf(b) << 16);
}
__device__ __forceinline__ int imin(int a, int b) { return a < b ? a : b; }
__device__ __forceinline__ int imax(int a, int b) { return a > b ? a : b; }

#define GM_CONV3   0   // B[k][n] = im2col 3x3 pad1 from f32 src0, k = cin*9 + r
#define GM_CONV1   1   // B[k][n] = src0[b][k][p]            (1x1 conv)
#define GM_CONCAT1 2   // 1x1 over concat(src0[256], src1)   (k<256 -> src0)
#define GM_VALBF   3   // B[k][n] = bf16 srcbf[b][k][p]      (deform-conv GEMM)

// C[M,N] = W[M,K] * B[K,N] + bias.  M = Cout, N = 16384 (= 4*64*64).
// Block: 256 thr (8 wave32). Tile 64(M) x 128(N), K step 32, double-buffered.
// Waves in 2(M) x 4(N) grid; each wave: 32x32 C tile -> 4 WMMA accumulators.
template <int MODE>
__global__ __launch_bounds__(256) void wmma_conv_gemm(
    const float* __restrict__ Wt,            // [Cout, K] f32
    const float* __restrict__ bias,          // [Cout]
    const float* __restrict__ src0,          // f32 activations
    const float* __restrict__ src1,          // second f32 source (concat)
    const unsigned short* __restrict__ srcbf,// bf16 source (GM_VALBF)
    float* __restrict__ out,                 // [4, Cout, 64, 64]
    int Cout, int K, int Cin) {
  constexpr int H = 64, Wd = 64, HW = 4096;

  __shared__ __align__(16) unsigned short Asm[2][64 * 40];    // 2 x  5 KB
  __shared__ __align__(16) unsigned short Bsm[2][128 * 40];   // 2 x 10 KB

  const int tid  = threadIdx.x;
  const int m0   = blockIdx.y * 64;
  const int n0   = blockIdx.x * 128;
  const int wave = tid >> 5;
  const int lane = tid & 31;
  const int wm   = wave >> 2;        // 0..1  (M, 32 rows each)
  const int wn   = wave & 3;         // 0..3  (N, 32 cols each)

  v8f c00 = {}, c01 = {}, c10 = {}, c11 = {};

  // A staging: 64x32 elems, 8/thread.  B staging: 128x32 elems, 16/thread.
  const int lrA = tid >> 2;          // 0..63
  const int lcA = (tid & 3) << 3;    // 0,8,16,24
  const int lrB = tid >> 1;          // 0..127
  const int lcB = (tid & 1) << 4;    // 0,16

  // decode this thread's B column (fixed across K loop); all shifts/masks
  const int nB  = n0 + lrB;
  const int bb  = nB >> 12;          // / 4096
  const int pp  = nB & 4095;
  const int yy  = pp >> 6;
  const int xx  = pp & 63;

  const int mA  = m0 + lrA;
  const unsigned aRow = (unsigned)imin(mA, Cout - 1) * (unsigned)K;

  auto stage = [&](int k0, int bufi) {
    // ---- A (weights), packed b128 store ----
    {
      const float* ap = Wt + aRow + (unsigned)(k0 + lcA);
      float va[8];
#pragma unroll
      for (int j = 0; j < 8; ++j) va[j] = ap[j];
      if (mA >= Cout) {
#pragma unroll
        for (int j = 0; j < 8; ++j) va[j] = 0.0f;
      }
      uint4 q;
      q.x = pack2(va[0], va[1]); q.y = pack2(va[2], va[3]);
      q.z = pack2(va[4], va[5]); q.w = pack2(va[6], va[7]);
      *reinterpret_cast<uint4*>(&Asm[bufi][lrA * 40 + lcA]) = q;
    }
    // ---- B (im2col, N-major), branch-free, packed b128 stores ----
#pragma unroll
    for (int h16 = 0; h16 < 2; ++h16) {
      const int kbase = k0 + lcB + h16 * 8;
      uint4 q;
      if (MODE == GM_VALBF) {
        unsigned short hv[8];
        const unsigned short* sp =
            srcbf + ((unsigned)bb * (unsigned)K + (unsigned)kbase) * HW + pp;
#pragma unroll
        for (int j = 0; j < 8; ++j) hv[j] = sp[j * HW];
        q.x = (unsigned)hv[0] | ((unsigned)hv[1] << 16);
        q.y = (unsigned)hv[2] | ((unsigned)hv[3] << 16);
        q.z = (unsigned)hv[4] | ((unsigned)hv[5] << 16);
        q.w = (unsigned)hv[6] | ((unsigned)hv[7] << 16);
      } else {
        float v[8];
#pragma unroll
        for (int j = 0; j < 8; ++j) {
          const int k = kbase + j;
          if (MODE == GM_CONV1) {
            v[j] = src0[((unsigned)(bb * K + k) << 12) + pp];
          } else if (MODE == GM_CONCAT1) {
            v[j] = (k < 256)
                       ? src0[((unsigned)(bb * 256 + k) << 12) + pp]
                       : src1[((unsigned)(bb * 256 + (k - 256)) << 12) + pp];
          } else {  // GM_CONV3: clamp address, select-zero
            const int cin = k / 9, r = k - cin * 9;   // const-9 magic mul
            const int sy = yy + r / 3 - 1;
            const int sx = xx + (r % 3) - 1;
            const bool ok = (sy >= 0) & (sy < H) & (sx >= 0) & (sx < Wd);
            const int cy = imin(imax(sy, 0), H - 1);
            const int cx = imin(imax(sx, 0), Wd - 1);
            const float t =
                src0[(((unsigned)(bb * Cin + cin) << 6) + cy) * 64u + cx];
            v[j] = ok ? t : 0.0f;
          }
        }
        q.x = pack2(v[0], v[1]); q.y = pack2(v[2], v[3]);
        q.z = pack2(v[4], v[5]); q.w = pack2(v[6], v[7]);
      }
      *reinterpret_cast<uint4*>(&Bsm[bufi][lrB * 40 + lcB + h16 * 8]) = q;
    }
  };

  const int nIter = K >> 5;
  stage(0, 0);

  const int half = (lane < 16) ? 0 : 8;
  const unsigned aOff = (unsigned)((wm * 32 + (lane & 15)) * 40 + half);
  const unsigned bOff = (unsigned)((wn * 32 + (lane & 15)) * 40 + half);

  for (int it = 0; it < nIter; ++it) {
    __syncthreads();
    const int cur = it & 1;
    if (it + 1 < nIter) stage((it + 1) << 5, cur ^ 1);

    union Frag { v16bf v; uint4 q[2]; };
    Frag a0, a1, b0, b1;
    {
      const unsigned short* pa = &Asm[cur][aOff];
      a0.q[0] = *reinterpret_cast<const uint4*>(pa);
      a0.q[1] = *reinterpret_cast<const uint4*>(pa + 16);
      a1.q[0] = *reinterpret_cast<const uint4*>(pa + 16 * 40);
      a1.q[1] = *reinterpret_cast<const uint4*>(pa + 16 * 40 + 16);
      const unsigned short* pb = &Bsm[cur][bOff];
      b0.q[0] = *reinterpret_cast<const uint4*>(pb);
      b0.q[1] = *reinterpret_cast<const uint4*>(pb + 16);
      b1.q[0] = *reinterpret_cast<const uint4*>(pb + 16 * 40);
      b1.q[1] = *reinterpret_cast<const uint4*>(pb + 16 * 40 + 16);
    }
    c00 = __builtin_amdgcn_wmma_f32_16x16x32_bf16(false, a0.v, false, b0.v,
                                                  (short)0, c00, false, false);
    c01 = __builtin_amdgcn_wmma_f32_16x16x32_bf16(false, a0.v, false, b1.v,
                                                  (short)0, c01, false, false);
    c10 = __builtin_amdgcn_wmma_f32_16x16x32_bf16(false, a1.v, false, b0.v,
                                                  (short)0, c10, false, false);
    c11 = __builtin_amdgcn_wmma_f32_16x16x32_bf16(false, a1.v, false, b1.v,
                                                  (short)0, c11, false, false);
  }

  // ---- epilogue: D layout (VGPR v: lanes 0-15 -> M=v, lanes 16-31 -> M=v+8)
  const int mrow0 = m0 + wm * 32 + ((lane < 16) ? 0 : 8);
  const int ncol0 = n0 + wn * 32 + (lane & 15);
#pragma unroll
  for (int mi = 0; mi < 2; ++mi) {
#pragma unroll
    for (int v = 0; v < 8; ++v) {
      const int m = mrow0 + mi * 16 + v;
      if (m < Cout) {
        const float bv = bias[m];
#pragma unroll
        for (int ni = 0; ni < 2; ++ni) {
          const int n = ncol0 + ni * 16;
          const int b2 = n >> 12, p2 = n & 4095;
          const v8f& acc =
              (mi == 0) ? (ni == 0 ? c00 : c01) : (ni == 0 ? c10 : c11);
          out[(((unsigned)(b2 * Cout + m)) << 12) + p2] = acc[v] + bv;
        }
      }
    }
  }
}

// --- bilinear 2x down 128->64 (align_corners=False => 2x2 average) ---
__global__ void k_down2x(const float* __restrict__ in, float* __restrict__ out) {
  const unsigned idx = blockIdx.x * 256u + threadIdx.x;   // < 4*256*64*64
  const unsigned x  = idx & 63;
  const unsigned y  = (idx >> 6) & 63;
  const unsigned bc = idx >> 12;
  const float* p = in + (bc << 14) + (y << 8) + (x << 1);
  out[idx] = 0.25f * (p[0] + p[1] + p[128] + p[129]);
}

// --- GroupNorm statistics: one block per (b, group); C=256, groups=32 ---
__global__ __launch_bounds__(256) void k_gn_stats(
    const float* __restrict__ x, float* __restrict__ mean,
    float* __restrict__ rstd) {
  const unsigned bg = blockIdx.x;                 // b*32 + g
  const unsigned base = bg << 15;                 // (b*256 + g*8)*4096
  float s = 0.f, ss = 0.f;
  for (unsigned i = threadIdx.x; i < 32768u; i += 256u) {
    const float v = x[base + i];
    s += v; ss += v * v;
  }
  __shared__ float sh[256], sh2[256];
  sh[threadIdx.x] = s; sh2[threadIdx.x] = ss;
  __syncthreads();
  for (int off = 128; off > 0; off >>= 1) {
    if ((int)threadIdx.x < off) {
      sh[threadIdx.x]  += sh[threadIdx.x + off];
      sh2[threadIdx.x] += sh2[threadIdx.x + off];
    }
    __syncthreads();
  }
  if (threadIdx.x == 0) {
    const float mu  = sh[0] * (1.f / 32768.f);
    const float var = sh2[0] * (1.f / 32768.f) - mu * mu;
    mean[bg] = mu;
    rstd[bg] = rsqrtf(var + 1e-6f);
  }
}

// --- GroupNorm apply + SiLU, in place ---
__global__ void k_gn_silu(float* __restrict__ x, const float* __restrict__ gamma,
                          const float* __restrict__ beta,
                          const float* __restrict__ mean,
                          const float* __restrict__ rstd) {
  const unsigned idx = blockIdx.x * 256u + threadIdx.x;   // < 4*256*4096
  const unsigned c  = (idx >> 12) & 255;
  const unsigned bg = ((idx >> 20) << 5) + (c >> 3);      // b*32 + c/8
  const float v  = x[idx];
  const float nv = (v - mean[bg]) * rstd[bg] * gamma[c] + beta[c];
  x[idx] = nv / (1.f + __expf(-nv));                      // silu
}

// --- depthwise 7x7, pad 3 ---
__global__ void k_dw7(const float* __restrict__ in, const float* __restrict__ w,
                      const float* __restrict__ bias, float* __restrict__ out) {
  const unsigned idx = blockIdx.x * 256u + threadIdx.x;   // < 4*256*4096
  const int x = (int)(idx & 63);
  const int y = (int)((idx >> 6) & 63);
  const unsigned c  = (idx >> 12) & 255;
  const unsigned bc = idx >> 12;
  const float* ip = in + (bc << 12);
  const float* wp = w + c * 49;
  float acc = bias[c];
#pragma unroll
  for (int i = 0; i < 7; ++i) {
    const int sy = y + i - 3;
    if (sy < 0 || sy >= 64) continue;
#pragma unroll
    for (int j = 0; j < 7; ++j) {
      const int sx = x + j - 3;
      if (sx < 0 || sx >= 64) continue;
      acc += ip[(sy << 6) + sx] * wp[i * 7 + j];
    }
  }
  out[idx] = acc;
}

// --- deformable sampling: block.y encodes (b,g,k); threads cover p ---
// writes bf16 val[b][(g*32+c)*9 + k][p]
__global__ void k_dcn_sample(const float* __restrict__ x,
                             const float* __restrict__ co,
                             unsigned short* __restrict__ val) {
  const int by = blockIdx.y;            // 0..287 = b*72 + g*9 + k
  const int k = by % 9;
  const int g = (by / 9) & 7;
  const int b = by / 72;
  const int p = (int)(blockIdx.x * 256u + threadIdx.x);   // 0..4095
  const int y = p >> 6, xx = p & 63;
  const int q = g * 9 + k;

  const float* cob = co + ((unsigned)b * 216u << 12);
  const float dy = cob[((unsigned)(2 * q) << 12) + p];
  const float dx = cob[((unsigned)(2 * q + 1) << 12) + p];
  float m = cob[((unsigned)(144 + q) << 12) + p];
  m = 1.f / (1.f + __expf(-m));         // sigmoid(mask)

  const float py = (float)(y - 1 + k / 3) + dy;
  const float px = (float)(xx - 1 + k % 3) + dx;
  const float fy = floorf(py), fx = floorf(px);
  const int y0 = (int)fy, x0 = (int)fx;
  const float wy = py - fy, wx = px - fx;

  const bool vy0 = (y0 >= 0 && y0 < 64);
  const bool vy1 = (y0 + 1 >= 0 && y0 + 1 < 64);
  const bool vx0 = (x0 >= 0 && x0 < 64);
  const bool vx1 = (x0 + 1 >= 0 && x0 + 1 < 64);
  const float a00 = (vy0 && vx0) ? (1.f - wy) * (1.f - wx) * m : 0.f;
  const float a01 = (vy0 && vx1) ? (1.f - wy) * wx * m : 0.f;
  const float a10 = (vy1 && vx0) ? wy * (1.f - wx) * m : 0.f;
  const float a11 = (vy1 && vx1) ? wy * wx * m : 0.f;
  const int cy0 = imin(imax(y0, 0), 63);
  const int cy1 = imin(imax(y0 + 1, 0), 63);
  const int cx0 = imin(imax(x0, 0), 63);
  const int cx1 = imin(imax(x0 + 1, 0), 63);
  const int l00 = (cy0 << 6) + cx0, l01 = (cy0 << 6) + cx1;
  const int l10 = (cy1 << 6) + cx0, l11 = (cy1 << 6) + cx1;

  const float* xb = x + (((unsigned)b * 256u + (unsigned)g * 32u) << 12);
  unsigned short* vb =
      val + (((unsigned)b * 2304u + (unsigned)g * 32u * 9u + (unsigned)k) << 12) + p;
#pragma unroll 4
  for (int c = 0; c < 32; ++c) {
    const float* xc = xb + ((unsigned)c << 12);
    const float v = a00 * xc[l00] + a01 * xc[l01] + a10 * xc[l10] + a11 * xc[l11];
    vb[(unsigned)c * 9u << 12] = f2bf(v);
  }
}

extern "C" void kernel_launch(void* const* d_in, const int* in_sizes, int n_in,
                              void* d_out, int out_size, void* d_ws,
                              size_t ws_size, hipStream_t stream) {
  (void)in_sizes; (void)n_in; (void)out_size; (void)ws_size;
  const float* x_main = (const float*)d_in[0];
  const float* prior  = (const float*)d_in[1];
  const float* ds_w = (const float*)d_in[2];   const float* ds_b = (const float*)d_in[3];
  const float* w1a  = (const float*)d_in[4];   const float* b1a  = (const float*)d_in[5];
  const float* g1a  = (const float*)d_in[6];   const float* be1a = (const float*)d_in[7];
  const float* w1b  = (const float*)d_in[8];   const float* b1b  = (const float*)d_in[9];
  const float* g1b  = (const float*)d_in[10];  const float* be1b = (const float*)d_in[11];
  const float* w1c  = (const float*)d_in[12];  const float* b1c  = (const float*)d_in[13];
  const float* w2   = (const float*)d_in[14];  const float* b2   = (const float*)d_in[15];
  const float* g2   = (const float*)d_in[16];  const float* be2  = (const float*)d_in[17];
  const float* co_w = (const float*)d_in[18];  const float* co_b = (const float*)d_in[19];
  const float* dcn_w = (const float*)d_in[20]; const float* dcn_b = (const float*)d_in[21];

  const size_t tens = (size_t)4 * 256 * 4096;   // 4,194,304

  // workspace carve-up
  float* W0    = (float*)d_ws;                    // 16.78 MB
  float* W1    = W0 + tens;                       // 16.78 MB
  float* Wco   = W1 + tens;                       // 14.16 MB (4*216*4096)
  float* stats = Wco + (size_t)4 * 216 * 4096;    // 256 floats
  float* mean  = stats;
  float* rstd  = stats + 128;
  unsigned short* Wval = (unsigned short*)(stats + 256);  // 75.5 MB bf16

  float* out_warp = (float*)d_out;
  float* out_feat = out_warp + tens;

  const dim3 blk(256);
  const dim3 gGemm(16384 / 128, 4);               // 128 x 4 (Cout<=256)
  const unsigned gElem = (unsigned)(tens / 256);  // 16384 blocks

  // 1) bilinear 2x down of prior -> W0
  k_down2x<<<gElem, blk, 0, stream>>>(prior, W0);

  // 2) 3x3 conv (ds): W0 -> W1
  wmma_conv_gemm<GM_CONV3><<<gGemm, blk, 0, stream>>>(
      ds_w, ds_b, W0, nullptr, nullptr, W1, 256, 2304, 256);

  // 3) 1x1 conv over concat(W1, x_main): -> W0
  wmma_conv_gemm<GM_CONCAT1><<<gGemm, blk, 0, stream>>>(
      w1a, b1a, W1, x_main, nullptr, W0, 256, 512, 512);

  // 4) GN + SiLU in place on W0
  k_gn_stats<<<128, blk, 0, stream>>>(W0, mean, rstd);
  k_gn_silu<<<gElem, blk, 0, stream>>>(W0, g1a, be1a, mean, rstd);

  // 5) depthwise 7x7: W0 -> W1
  k_dw7<<<gElem, blk, 0, stream>>>(W0, w1b, b1b, W1);

  // 6) GN + SiLU in place on W1
  k_gn_stats<<<128, blk, 0, stream>>>(W1, mean, rstd);
  k_gn_silu<<<gElem, blk, 0, stream>>>(W1, g1b, be1b, mean, rstd);

  // 7) 1x1 conv (w1c): W1 -> W0
  wmma_conv_gemm<GM_CONV1><<<gGemm, blk, 0, stream>>>(
      w1c, b1c, W1, nullptr, nullptr, W0, 256, 256, 256);

  // 8) 3x3 conv (w2): W0 -> out_feat
  wmma_conv_gemm<GM_CONV3><<<gGemm, blk, 0, stream>>>(
      w2, b2, W0, nullptr, nullptr, out_feat, 256, 2304, 256);

  // 9) GN + SiLU in place on out_feat  => offset_feat (second output)
  k_gn_stats<<<128, blk, 0, stream>>>(out_feat, mean, rstd);
  k_gn_silu<<<gElem, blk, 0, stream>>>(out_feat, g2, be2, mean, rstd);

  // 10) 3x3 conv (co): out_feat -> Wco (216 ch: offsets + mask logits)
  wmma_conv_gemm<GM_CONV3><<<gGemm, blk, 0, stream>>>(
      co_w, co_b, out_feat, nullptr, nullptr, Wco, 216, 2304, 256);

  // 11) deformable bilinear sampling: (x_main, Wco) -> Wval (bf16 [4,2304,4096])
  k_dcn_sample<<<dim3(16, 288), blk, 0, stream>>>(x_main, Wco, Wval);

  // 12) deform-conv GEMM: dcn_w [256,2304] x Wval -> out_warp (first output)
  wmma_conv_gemm<GM_VALBF><<<gGemm, blk, 0, stream>>>(
      dcn_w, dcn_b, nullptr, nullptr, Wval, out_warp, 256, 2304, 256);
}
